// pCosineSim_23201413333422
// MI455X (gfx1250) — matrix operand order repaired and checked by reading, not compiled, not run on hardware
//
#include <hip/hip_runtime.h>
#include <hip/hip_bf16.h>

typedef _Float16 v8h  __attribute__((ext_vector_type(8)));
typedef _Float16 v16h __attribute__((ext_vector_type(16)));
typedef float    v8f  __attribute__((ext_vector_type(8)));

#define NROWS 8192
#define DCOLS 256
#define KC    32          // K-chunk per stage == one WMMA K step
#define LDK   40          // KC + 8 halfs padding (80B row stride, conflict-free)
#define NCHUNK (DCOLS / KC)
#define EPSV  1e-6f

static __device__ __forceinline__ v16h cat16(v8h a, v8h b) {
    return __builtin_shufflevector(a, b, 0,1,2,3,4,5,6,7,8,9,10,11,12,13,14,15);
}

// ---------------------------------------------------------------------------
// Prep: per-row rsqrt(sum(x^2)), scale, split into f16 hi/lo pair.
// One block per row, one thread per column.
// ---------------------------------------------------------------------------
__global__ __launch_bounds__(256) void cosim_prep(const float* __restrict__ x,
                                                  _Float16* __restrict__ xhi,
                                                  _Float16* __restrict__ xlo) {
    __shared__ float red[256];
    const int row = blockIdx.x;
    const int t   = threadIdx.x;
    const float v = x[(size_t)row * DCOLS + t];
    red[t] = v * v;
    __syncthreads();
    #pragma unroll
    for (int s = 128; s > 0; s >>= 1) {
        if (t < s) red[t] += red[t + s];
        __syncthreads();
    }
    const float rn = rsqrtf(red[0]);
    const float y  = v * rn;
    const _Float16 hi = (_Float16)y;
    const float lo = y - (float)hi;
    xhi[(size_t)row * DCOLS + t] = hi;
    xlo[(size_t)row * DCOLS + t] = (_Float16)lo;
}

// ---------------------------------------------------------------------------
// GEMM: lower-triangular 128x128 blocks of Y * Y^T with f16 hi/lo split
// (hi*hi + hi*lo + lo*hi -> ~fp32 accuracy), mirrored store for symmetry.
// 256 threads = 8 waves; wave grid 4(M) x 2(N); each wave: 2x4 tiles of 16x16.
// ---------------------------------------------------------------------------
__global__ __launch_bounds__(256) void cosim_gemm(const _Float16* __restrict__ xhi,
                                                  const _Float16* __restrict__ xlo,
                                                  float* __restrict__ out) {
    __shared__ _Float16 sAhi[128 * LDK];
    __shared__ _Float16 sAlo[128 * LDK];
    __shared__ _Float16 sBhi[128 * LDK];
    __shared__ _Float16 sBlo[128 * LDK];

    // triangular block decode: t = bi*(bi+1)/2 + bj, bi >= bj
    const int t = blockIdx.x;
    int bi = (int)((sqrtf(8.0f * (float)t + 1.0f) - 1.0f) * 0.5f);
    while ((bi + 1) * (bi + 2) / 2 <= t) ++bi;
    while (bi * (bi + 1) / 2 > t) --bi;
    const int bj = t - bi * (bi + 1) / 2;

    const int rowbase = bi * 128;   // M block
    const int colbase = bj * 128;   // N block

    const int tid  = threadIdx.x;
    const int lane = tid & 31;
    const int wave = tid >> 5;
    const int wm   = wave & 3;      // 4 waves along M (32 rows each)
    const int wn   = wave >> 2;     // 2 waves along N (64 cols each)
    const int lg   = lane >> 4;     // lane group 0/1
    const int ll   = lane & 15;

    // staging: each thread copies 16 halfs (two b128) per array per chunk
    const int lr = tid >> 1;            // 0..127
    const int lc = (tid & 1) << 4;      // 0 or 16
    const size_t gA = (size_t)(rowbase + lr) * DCOLS + lc;
    const size_t gB = (size_t)(colbase + lr) * DCOLS + lc;
    const int sIdx = lr * LDK + lc;

    v8f acc[2][4] = {};

    for (int ch = 0; ch < NCHUNK; ++ch) {
        const size_t ko = (size_t)ch * KC;
        *(v8h*)&sAhi[sIdx]     = *(const v8h*)(xhi + gA + ko);
        *(v8h*)&sAhi[sIdx + 8] = *(const v8h*)(xhi + gA + ko + 8);
        *(v8h*)&sAlo[sIdx]     = *(const v8h*)(xlo + gA + ko);
        *(v8h*)&sAlo[sIdx + 8] = *(const v8h*)(xlo + gA + ko + 8);
        *(v8h*)&sBhi[sIdx]     = *(const v8h*)(xhi + gB + ko);
        *(v8h*)&sBhi[sIdx + 8] = *(const v8h*)(xhi + gB + ko + 8);
        *(v8h*)&sBlo[sIdx]     = *(const v8h*)(xlo + gB + ko);
        *(v8h*)&sBlo[sIdx + 8] = *(const v8h*)(xlo + gB + ko + 8);
        if (ch + 1 < NCHUNK) {              // -> global_prefetch_b8
            __builtin_prefetch(xhi + gA + ko + KC, 0, 1);
            __builtin_prefetch(xhi + gB + ko + KC, 0, 1);
        }
        __syncthreads();

        #pragma unroll
        for (int i = 0; i < 2; ++i) {
            // A fragment 16x32 f16: VGPR0-3 = K[lg*8 .. +8), VGPR4-7 = K[lg*8+16 .. +8)
            const int arow = (wm * 32 + i * 16 + ll) * LDK + lg * 8;
            const v16h ahi = cat16(*(const v8h*)&sAhi[arow], *(const v8h*)&sAhi[arow + 16]);
            const v16h alo = cat16(*(const v8h*)&sAlo[arow], *(const v8h*)&sAlo[arow + 16]);
            #pragma unroll
            for (int j = 0; j < 4; ++j) {
                // B fragment 32x16 f16: VGPR0-3 = K[lg*8 .. +8), VGPR4-7 = K[lg*8+16 .. +8)
                const int brow = (wn * 64 + j * 16 + ll) * LDK + lg * 8;
                const v16h bhi = cat16(*(const v8h*)&sBhi[brow], *(const v8h*)&sBhi[brow + 16]);
                const v16h blo = cat16(*(const v8h*)&sBlo[brow], *(const v8h*)&sBlo[brow + 16]);
                acc[i][j] = __builtin_amdgcn_wmma_f32_16x16x32_f16(
                    false, ahi, false, bhi, (short)0, acc[i][j], false, false);
                acc[i][j] = __builtin_amdgcn_wmma_f32_16x16x32_f16(
                    false, ahi, false, blo, (short)0, acc[i][j], false, false);
                acc[i][j] = __builtin_amdgcn_wmma_f32_16x16x32_f16(
                    false, alo, false, bhi, (short)0, acc[i][j], false, false);
            }
        }
        __syncthreads();
    }

    // Epilogue: clamp, direct store at (bi,bj); mirrored store at (bj,bi).
    // C fragment: VGPR r, lane l -> element (M = r + 8*(l>>4), N = l&15).
    const int mlane = lg * 8;
    #pragma unroll
    for (int i = 0; i < 2; ++i) {
        const int m0 = rowbase + wm * 32 + i * 16;
        #pragma unroll
        for (int j = 0; j < 4; ++j) {
            const int n0 = colbase + wn * 64 + j * 16;
            v8f c = acc[i][j];
            #pragma unroll
            for (int r = 0; r < 8; ++r) c[r] = fmaxf(c[r], EPSV);

            float* pd = out + (size_t)(m0 + mlane) * NROWS + n0 + ll;
            #pragma unroll
            for (int r = 0; r < 8; ++r) pd[(size_t)r * NROWS] = c[r];

            if (bi != bj) {
                // transpose: per lane, M runs contiguously -> two float4 stores
                float* pt = out + (size_t)(n0 + ll) * NROWS + m0 + mlane;
                float4 f0 = {c[0], c[1], c[2], c[3]};
                float4 f1 = {c[4], c[5], c[6], c[7]};
                *(float4*)pt       = f0;
                *(float4*)(pt + 4) = f1;
            }
        }
    }
}

extern "C" void kernel_launch(void* const* d_in, const int* in_sizes, int n_in,
                              void* d_out, int out_size, void* d_ws, size_t ws_size,
                              hipStream_t stream) {
    const float* x = (const float*)d_in[0];
    float* out = (float*)d_out;
    _Float16* xhi = (_Float16*)d_ws;                       // 4 MB
    _Float16* xlo = xhi + (size_t)NROWS * DCOLS;           // 4 MB

    cosim_prep<<<NROWS, 256, 0, stream>>>(x, xhi, xlo);

    const int nblk = (NROWS / 128) * (NROWS / 128 + 1) / 2;  // 2080 triangular blocks
    cosim_gemm<<<nblk, 256, 0, stream>>>(xhi, xlo, out);
}
